// GraphConv_51969104281840
// MI455X (gfx1250) — matrix operand order repaired and checked by reading, not compiled, not run on hardware
//
#include <hip/hip_runtime.h>

// ---------------------------------------------------------------------------
// GraphConv for MI455X (gfx1250): out = relu( (I ++ A) * (h ⊗ V) )
// Stage 1 (k_project):  VWt[b][c][j*4+l] = sum_f V[b,j,f] * h[1+l,c,f]  (bf16)
// Stage 2 (k_aggregate): out[b,i,c] = relu( sum_k A[b,i,k]*VWt[b,c,k]
//                                         + sum_f V[b,i,f]*h[0,c,f] )
// K of the aggregation GEMM is N*L = 4096 == A's contiguous (j,l) layout, so
// A streams linearly from HBM (134 MB — the roofline term, ~6.5 us).
// Each aggregation wave computes one 16(i) x 32(c) tile: one A-fragment
// conversion (8x v_perm_b32) feeds two v_wmma_f32_16x16x32_bf16 issues.
// ---------------------------------------------------------------------------

typedef __attribute__((ext_vector_type(16))) __bf16 v16bf;
typedef __attribute__((ext_vector_type(8)))  float  v8f;

namespace {
constexpr int kB = 8, kN = 1024, kF = 256, kC = 256, kL = 4;
constexpr int kK = kN * kL;            // 4096 contraction length (j,l)
constexpr int kCT = kC / 16;           // 16 c-tiles
constexpr int kNT = kN / 16;           // 64 i/j-tiles
}

union FragBF {
    unsigned u[8];
    v16bf    v;
};

// Pack two f32 into one dword of bf16 (lo -> [15:0], hi -> [31:16]).
#if __has_builtin(__builtin_amdgcn_cvt_pk_bf16_f32)
__device__ __forceinline__ unsigned pack2(float lo, float hi) {
    auto r = __builtin_amdgcn_cvt_pk_bf16_f32(lo, hi);  // 1 VALU, RNE
    return __builtin_bit_cast(unsigned, r);
}
#elif __has_builtin(__builtin_amdgcn_perm)
__device__ __forceinline__ unsigned pack2(float lo, float hi) {
    // v_perm_b32: byte pool = {src0=hi: bytes 4..7, src1=lo: bytes 0..3}
    // select (2,3,6,7) -> { hi[31:16], lo[31:16] } : 1 VALU truncating pack
    return __builtin_amdgcn_perm(__float_as_uint(hi), __float_as_uint(lo),
                                 0x07060302u);
}
#else
__device__ __forceinline__ unsigned pack2(float lo, float hi) {
    return (__float_as_uint(lo) >> 16) | (__float_as_uint(hi) & 0xFFFF0000u);
}
#endif

// Build a 16-element bf16 fragment (8 VGPRs) from 4 float4 chunks in K order.
__device__ __forceinline__ void cvt_frag(FragBF& f, float4 a0, float4 a1,
                                         float4 a2, float4 a3) {
    f.u[0] = pack2(a0.x, a0.y); f.u[1] = pack2(a0.z, a0.w);
    f.u[2] = pack2(a1.x, a1.y); f.u[3] = pack2(a1.z, a1.w);
    f.u[4] = pack2(a2.x, a2.y); f.u[5] = pack2(a2.z, a2.w);
    f.u[6] = pack2(a3.x, a3.y); f.u[7] = pack2(a3.z, a3.w);
}

__device__ __forceinline__ void load_bf16_frag(FragBF& f,
                                               const unsigned short* p) {
    const uint4* q = (const uint4*)p;          // 32B-aligned by construction
    uint4 w0 = q[0], w1 = q[1];
    f.u[0] = w0.x; f.u[1] = w0.y; f.u[2] = w0.z; f.u[3] = w0.w;
    f.u[4] = w1.x; f.u[5] = w1.y; f.u[6] = w1.z; f.u[7] = w1.w;
}

__device__ __forceinline__ v8f wmma_bf16(const FragBF& a, const FragBF& b,
                                         v8f c) {
    return __builtin_amdgcn_wmma_f32_16x16x32_bf16(
        false, a.v, false, b.v, (short)0, c, false, false);
}

// ---------------------------------------------------------------------------
// Kernel 1: projection GEMM per (b,l): M=c (h rows), N=j (V rows), K=f.
// D tile stored transposed+interleaved into VWt[b][c][j*kL + l] as bf16.
// ---------------------------------------------------------------------------
__global__ __launch_bounds__(256) void k_project(
    const float* __restrict__ V,       // (B,N,F) f32
    const float* __restrict__ H,       // (L+1,C,F) f32
    unsigned short* __restrict__ VWt)  // (B,C,N*L) bf16 scratch
{
    const int wave = (blockIdx.x * 256 + threadIdx.x) >> 5;
    const int lane = threadIdx.x & 31;
    const int half = lane >> 4;        // K-offset group per ISA fragment layout
    const int lr   = lane & 15;

    int t = wave;
    const int b  = t / (kL * kCT * kNT); t -= b * (kL * kCT * kNT);
    const int l  = t / (kCT * kNT);      t -= l * (kCT * kNT);
    const int c0 = (t / kNT) * 16;
    const int j0 = (t % kNT) * 16;

    const float* hrow = H + ((size_t)(l + 1) * kC + (c0 + lr)) * kF;  // A-op row
    const float* vrow = V + ((size_t)b * kN + (j0 + lr)) * kF;        // B-op col

    v8f acc = {};
    for (int f0 = 0; f0 < kF; f0 += 32) {
        // A fragment: 16 contiguous K starting at f0 + half*16
        const float4* pa  = (const float4*)(hrow + f0 + half * 16);
        const float4* pa2 = (const float4*)(hrow + f0 + half * 16 + 8);
        FragBF fa; cvt_frag(fa, pa[0], pa[1], pa2[0], pa2[1]);
        // B fragment: 16 contiguous K starting at f0 + half*16
        const float4* pb  = (const float4*)(vrow + f0 + half * 16);
        FragBF fb; cvt_frag(fb, pb[0], pb[1], pb[2], pb[3]);
        acc = wmma_bf16(fa, fb, acc);
    }

    // D layout: VGPR r -> M = r + 8*half, N = lane&15. Scatter bf16 stores.
    unsigned short* base = VWt + (size_t)b * kC * kK;
    const int jcol = j0 + lr;
#pragma unroll
    for (int r = 0; r < 8; ++r) {
        const int c = c0 + r + 8 * half;
        base[(size_t)c * kK + (size_t)jcol * kL + l] =
            (unsigned short)(__float_as_uint(acc[r]) >> 16);
    }
}

// ---------------------------------------------------------------------------
// Kernel 2: aggregation. One block = one (b, i0); its 8 waves cover the 8
// 32-wide c-groups, so the block's 16 A rows are shared via WGP$/L2.
// Each wave: 16x32 output tile, two accumulators, one A-frag -> two WMMAs.
// ---------------------------------------------------------------------------
__global__ __launch_bounds__(256) void k_aggregate(
    const float* __restrict__ A,             // (B,N,N,L) f32
    const float* __restrict__ V,             // (B,N,F) f32
    const float* __restrict__ H,             // (L+1,C,F) f32 (l=0 slab used)
    const unsigned short* __restrict__ VWt,  // (B,C,N*L) bf16
    float* __restrict__ Out)                 // (B,N,C) f32
{
    const int wave = threadIdx.x >> 5;       // 0..7 -> c-group
    const int lane = threadIdx.x & 31;
    const int half = lane >> 4;
    const int lr   = lane & 15;

    const int b  = blockIdx.x / kNT;
    const int i0 = (blockIdx.x % kNT) * 16;
    const int c0 = wave * 32;                // this wave's two c-tiles

    const float*          arow  = A   + ((size_t)b * kN + (i0 + lr)) * kK;
    const unsigned short* wrow0 = VWt + ((size_t)b * kC + (c0 + lr)) * kK;
    const unsigned short* wrow1 = VWt + ((size_t)b * kC + (c0 + 16 + lr)) * kK;

    v8f acc0 = {}, acc1 = {};
    for (int k0 = 0; k0 < kK; k0 += 32) {
        // A fragment from f32 adjacency row (two contiguous 32B chunks/lane)
        const float4* pa  = (const float4*)(arow + k0 + half * 16);
        const float4* pa2 = (const float4*)(arow + k0 + half * 16 + 8);
        FragBF fa; cvt_frag(fa, pa[0], pa[1], pa2[0], pa2[1]);
        // Two B fragments: pre-converted bf16, contiguous 32B runs
        FragBF fb0, fb1;
        load_bf16_frag(fb0, wrow0 + k0 + half * 16);
        load_bf16_frag(fb1, wrow1 + k0 + half * 16);
        // keep the HBM stream of A ahead (~2KB / 16 iterations per lane-row)
        __builtin_prefetch(arow + k0 + 512, 0, 1);
        acc0 = wmma_bf16(fa, fb0, acc0);
        acc1 = wmma_bf16(fa, fb1, acc1);
    }

    // Fused identity edge: out += V[b,i,:] . h[0,c,:]
    const float* vrow  = V + ((size_t)b * kN + (i0 + lr)) * kF;
    const float* hrow0 = H + (size_t)(c0 + lr) * kF;        // h_weights[0]
    const float* hrow1 = H + (size_t)(c0 + 16 + lr) * kF;
    for (int f0 = 0; f0 < kF; f0 += 32) {
        const float4* pa  = (const float4*)(vrow + f0 + half * 16);
        const float4* pa2 = (const float4*)(vrow + f0 + half * 16 + 8);
        FragBF fa; cvt_frag(fa, pa[0], pa[1], pa2[0], pa2[1]);
        const float4* pb0 = (const float4*)(hrow0 + f0 + half * 16);
        const float4* pb1 = (const float4*)(hrow1 + f0 + half * 16);
        FragBF fb0; cvt_frag(fb0, pb0[0], pb0[1], pb0[2], pb0[3]);
        FragBF fb1; cvt_frag(fb1, pb1[0], pb1[1], pb1[2], pb1[3]);
        acc0 = wmma_bf16(fa, fb0, acc0);
        acc1 = wmma_bf16(fa, fb1, acc1);
    }

    // ReLU + store; each half-wave writes contiguous 64B row segments
#pragma unroll
    for (int r = 0; r < 8; ++r) {
        const int i = i0 + r + 8 * half;
        float x0 = acc0[r];
        float x1 = acc1[r];
        float* orow = Out + ((size_t)b * kN + i) * kC;
        orow[c0 + lr]      = x0 > 0.f ? x0 : 0.f;
        orow[c0 + 16 + lr] = x1 > 0.f ? x1 : 0.f;
    }
}

extern "C" void kernel_launch(void* const* d_in, const int* in_sizes, int n_in,
                              void* d_out, int out_size, void* d_ws, size_t ws_size,
                              hipStream_t stream) {
    (void)in_sizes; (void)n_in; (void)out_size; (void)ws_size;
    const float* V = (const float*)d_in[0];   // (B,N,F)
    const float* A = (const float*)d_in[1];   // (B,N,N,L)
    const float* H = (const float*)d_in[2];   // (L+1,C,F)
    float* Out = (float*)d_out;
    unsigned short* VWt = (unsigned short*)d_ws;  // 16 MiB bf16 scratch

    // Stage 1: 8*4*16*64 = 32768 waves, 8 waves (256 threads) per block
    const int waves1 = kB * kL * kCT * kNT;
    k_project<<<dim3(waves1 / 8), dim3(256), 0, stream>>>(V, H, VWt);

    // Stage 2: one block per (b, i0): 8*64 = 512 blocks x 8 waves
    k_aggregate<<<dim3(kB * kNT), dim3(256), 0, stream>>>(A, V, H, VWt, Out);
}
